// FMNISTCanny_44126493999426
// MI455X (gfx1250) — matrix operand order repaired and checked by reading, not compiled, not run on hardware
//
#include <hip/hip_runtime.h>
#include <hip/hip_bf16.h>

typedef __attribute__((ext_vector_type(16))) _Float16 v16h;
typedef __attribute__((ext_vector_type(8)))  float    v8f;

#define GAMMA_C 0.005f
#define EPS_C   1e-9f

// One 16x16 f32 output tile of C = A(32x32,f16) x B(32x32,f16), K=32, via a
// single V_WMMA_F32_16X16X32_F16. Fragment layouts per CDNA5 ISA 7.12.2.
// B is supplied TRANSPOSED (Bt[n][k]) so every lane's 16 halves are two
// contiguous 16-byte runs in LDS (vectorizable to ds_load_b128), same as A.
__device__ __forceinline__ void wmma_tile32(const _Float16 (*A)[32],
                                            const _Float16 (*Bt)[32],
                                            float (*C)[32],
                                            int tile, int lane) {
  const int ti   = (tile >> 1) & 1;
  const int tj   = tile & 1;
  const int half = lane >> 4;   // lane group 0..15 vs 16..31
  const int l15  = lane & 15;
  const int am   = ti * 16 + l15;   // A row (M)
  const int bn   = tj * 16 + l15;   // B col (N)
  v16h a, b;
#pragma unroll
  for (int j = 0; j < 16; ++j) {
    // A 16x32 f16: j<8 -> K = half*8 + j ; j>=8 -> K = 16 + half*8 + (j-8)
    const int ka = (j & 7) + (half << 3) + ((j & 8) << 1);
    a[j] = A[am][ka];
    // B 32x16 f16: lane needs B[K = ka][bn] == Bt[bn][ka] (same K order as A)
    b[j] = Bt[bn][ka];
  }
  v8f c = {};
  c = __builtin_amdgcn_wmma_f32_16x16x32_f16(false, a, false, b, (short)0, c,
                                             false, false);
#pragma unroll
  for (int v = 0; v < 8; ++v)
    C[ti * 16 + (half << 3) + v][bn] = c[v];   // C/D: M = v + 8*half
}

__global__ void __launch_bounds__(1024)
fmnist_canny_kernel(const float* __restrict__ xin,
                    const float* __restrict__ msk,
                    const float* __restrict__ gk,   // 25 taps
                    float* __restrict__ out) {
  __shared__ _Float16 Gh[32][32];           // symmetric Toeplitz: A and Bt roles
  __shared__ _Float16 XhT[32][32], MhT[32][32];   // transposed inputs (B role)
  __shared__ _Float16 Txh[32][32], Tmh[32][32];   // stage-1 results (A role)
  __shared__ float F0[32][32], F1[32][32], F2[32][32], F3[32][32], F4[32][32];
  __shared__ unsigned int lowRow[32], highRow[32];

  const int tid  = threadIdx.x;
  const int r    = tid >> 5;       // each wave owns one image row
  const int c    = tid & 31;
  const int lane = tid & 31;
  const int wid  = tid >> 5;
  const bool act = (r < 28) && (c < 28);
  const int idx  = r * 28 + c;

  // ---- Build banded Toeplitz Gaussian G (symmetric) and transposed inputs ----
  {
    const int d = c - r;
    _Float16 gv = (_Float16)0.0f;
    if (act && d >= -12 && d <= 12) gv = (_Float16)gk[d + 12];
    Gh[r][c]  = gv;                                       // G[r][k] = g[k-r+12]
    XhT[r][c] = act ? (_Float16)xin[c * 28 + r] : (_Float16)0.0f;  // X^T
    MhT[r][c] = act ? (_Float16)msk[c * 28 + r] : (_Float16)0.0f;  // M^T
  }
  __syncthreads();

  // ---- Stage 1: F0 = G @ X, F1 = G @ M (smooth along H). 8 waves, 8 WMMAs ----
  if (wid < 8) {
    if (wid < 4) wmma_tile32(Gh, XhT, F0, wid, lane);
    else         wmma_tile32(Gh, MhT, F1, wid & 3, lane);
  }
  __syncthreads();
  Txh[r][c] = (_Float16)F0[r][c];
  Tmh[r][c] = (_Float16)F1[r][c];
  __syncthreads();

  // ---- Stage 2: F0 = Tx @ G, F1 = Tm @ G (smooth along W). G^T == G ----
  if (wid < 8) {
    if (wid < 4) wmma_tile32(Txh, Gh, F0, wid, lane);
    else         wmma_tile32(Tmh, Gh, F1, wid & 3, lane);
  }
  __syncthreads();

  // ---- xs = smooth(x) / (smooth(mask) + 1e-12) ----
  F2[r][c] = act ? (F0[r][c] / (F1[r][c] + 1e-12f)) : 0.0f;
  __syncthreads();

  // ---- Sobel: [-1,0,1] derivative (replicate pad) then [1,2,1] smooth ----
  {
    const int cp = (c < 27) ? c + 1 : 27, cm = (c > 0) ? c - 1 : 0;
    const int rp = (r < 27) ? r + 1 : 27, rm = (r > 0) ? r - 1 : 0;
    F0[r][c] = F2[r][cp] - F2[r][cm];   // dW
    F1[r][c] = F2[rp][c] - F2[rm][c];   // dH
  }
  __syncthreads();
  {
    const int cp = (c < 27) ? c + 1 : 27, cm = (c > 0) ? c - 1 : 0;
    const int rp = (r < 27) ? r + 1 : 27, rm = (r > 0) ? r - 1 : 0;
    F3[r][c] = F0[rm][c] + 2.0f * F0[r][c] + F0[rp][c];   // jsob
    F4[r][c] = F1[r][cm] + 2.0f * F1[r][c] + F1[r][cp];   // isob
  }
  __syncthreads();

  const float iv   = F4[r][c];
  const float jv   = F3[r][c];
  const float mag2 = iv * iv + jv * jv;
  const float m    = sqrtf(mag2 + EPS_C);   // THRES=0 gate is a no-op (m>0)
  F0[r][c] = act ? m : 0.0f;                // magnitude grid (zero padding)
  __syncthreads();

  // ---- NMS sectors + local maxima ----
  float s0 = 0.f, s1 = 0.f, s2 = 0.f, s3 = 0.f;
  bool lmv = false;
  bool lowv = false, highv = false;
  if (act) {
    const float ai = fabsf(iv), aj = fabsf(jv);
    // erosion of mask with full 3x3 structure, zero border (mask is M^T in LDS)
    bool er2 = true;
#pragma unroll
    for (int dr = -1; dr <= 1; ++dr)
#pragma unroll
      for (int dc = -1; dc <= 1; ++dc) {
        const int rr = r + dr, cc = c + dc;
        const bool mb = (rr >= 0 && rr < 28 && cc >= 0 && cc < 28) &&
                        ((float)MhT[cc][rr] != 0.0f);
        er2 = er2 && mb;
      }
    const bool er   = er2 && (mag2 > 0.0f);
    const bool same = ((iv >= 0.f) && (jv >= 0.f)) || ((iv <= 0.f) && (jv <= 0.f));
    const bool opp  = ((iv <= 0.f) && (jv >= 0.f)) || ((iv >= 0.f) && (jv <= 0.f));

    auto nbm = [&](int dr, int dc) -> float {
      const int rr = r + dr, cc = c + dc;
      return (rr >= 0 && rr < 28 && cc >= 0 && cc < 28) ? F0[rr][cc] : 0.0f;
    };
    auto sector = [&](bool pts, float w, float c1p, float c2p, float c1m,
                      float c2m, bool bug, float& s, bool& l) {
      const float ip = c2p * w + c1p * (1.0f - w);
      const float im = c2m * w + c1m * (1.0f - w);
      const float t1 = fmaxf(0.0f, GAMMA_C - m + ip);
      const float t2 = fmaxf(0.0f, GAMMA_C - m + (bug ? ip : im));
      s = pts ? fmaxf(t1, t2) : 0.0f;
      l = (ip <= m) && (im <= m);
    };

    bool l0, l1, l2, l3;
    const bool p0 = er && same && (ai >= aj);
    sector(p0, aj / (ai + EPS_C), nbm(1, 0), nbm(1, 1), nbm(-1, 0), nbm(-1, -1),
           false, s0, l0);
    const bool p1 = er && same && (ai <= aj);
    sector(p1, ai / (p1 ? aj : 1.0f), nbm(0, 1), nbm(1, 1), nbm(0, -1),
           nbm(-1, -1), false, s1, l1);
    const bool p2 = er && opp && (ai <= aj);   // reuses 'plus' interp (bug=true)
    sector(p2, ai / (p2 ? aj : 1.0f), nbm(0, 1), nbm(-1, 1), nbm(0, -1),
           nbm(1, -1), true, s2, l2);
    const bool p3 = er && opp && (ai >= aj);
    sector(p3, aj / (p3 ? ai : 1.0f), nbm(-1, 0), nbm(-1, 1), nbm(1, 0),
           nbm(1, -1), false, s3, l3);

    // sequential overwrite semantics of the reference
    if (p0) lmv = l0;
    if (p1) lmv = l1;
    if (p2) lmv = l2;
    if (p3) lmv = l3;

    highv = lmv && (m >= 0.2f);
    lowv  = lmv && (m >= 0.1f);

    out[idx]           = s0;
    out[784 + idx]     = s1;
    out[2 * 784 + idx] = s2;
    out[3 * 784 + idx] = s3;
    out[4 * 784 + idx] = lmv ? 1.0f : 0.0f;
    out[5 * 784 + idx] = 1.0f;              // 'test'
  }

  // ---- Hysteresis: pack each 28-pixel row into a 32-bit mask (one per wave),
  //      then run all 56 dependent dilations inside a single wave: row dilation
  //      is bit-shifts, vertical dilation is two lane shuffles. No barriers in
  //      the loop. final = ((dilate^56)(high)) & low per iteration.
  const unsigned int rowlow  = (unsigned int)__ballot(lowv);
  const unsigned int rowhigh = (unsigned int)__ballot(highv);
  if (lane == 0) {
    lowRow[wid]  = rowlow;    // rows >= 28 are all-zero
    highRow[wid] = rowhigh;
  }
  __syncthreads();
  if (wid == 0) {
    const unsigned int lo = lowRow[lane];
    unsigned int cur = highRow[lane];
    for (int it = 0; it < 56; ++it) {
      unsigned int up = __shfl_up(cur, 1);
      if (lane == 0) up = 0;
      unsigned int dn = __shfl_down(cur, 1);
      if (lane == 31) dn = 0;
      unsigned int comb = up | cur | dn;              // vertical 3-dilate
      comb = comb | (comb << 1) | (comb >> 1);        // horizontal 3-dilate
      cur = comb & lo;                                // & low (kills bit 28 too)
    }
    highRow[lane] = cur;   // final hysteresis mask, one word per row
  }
  __syncthreads();
  if (act)
    out[6 * 784 + idx] = ((highRow[r] >> c) & 1u) ? 1.0f : 0.0f;  // mag_out fwd
}

extern "C" void kernel_launch(void* const* d_in, const int* in_sizes, int n_in,
                              void* d_out, int out_size, void* d_ws,
                              size_t ws_size, hipStream_t stream) {
  (void)in_sizes; (void)n_in; (void)d_ws; (void)ws_size; (void)out_size;
  const float* x    = (const float*)d_in[0];
  const float* mask = (const float*)d_in[1];
  const float* gk   = (const float*)d_in[2];
  float* out        = (float*)d_out;
  fmnist_canny_kernel<<<1, 1024, 0, stream>>>(x, mask, gk, out);
}
